// Qwen3_5Attention_79010218377271
// MI455X (gfx1250) — matrix operand order repaired and checked by reading, not compiled, not run on hardware
//
#include <hip/hip_runtime.h>
#include <hip/hip_bf16.h>
#include <stdint.h>

typedef __attribute__((ext_vector_type(16))) _Float16 v16h;
typedef __attribute__((ext_vector_type(8)))  _Float16 v8h;
typedef __attribute__((ext_vector_type(8)))  float    v8f;
typedef __attribute__((ext_vector_type(4)))  unsigned int u32x4;
typedef __attribute__((ext_vector_type(4)))  int      i32x4;
typedef __attribute__((ext_vector_type(8)))  int      i32x8;

#define T_DIM   4096
#define HID_DIM 2048
#define NH      16
#define NKV     2
#define HD      128
#define NQKV    4608   // (2*16 + 2*2) * 128
#define QSZ     4096   // 2*H*D

#define BROW    40     // staged B row stride in halves (32 data + 8 TDM pad)

// ---------------------------------------------------------------------------
// WMMA helper: D = A(16x32 f16) * B(32x16 f16) + C(16x16 f32)
// ---------------------------------------------------------------------------
__device__ __forceinline__ v8f wmma_f16(v16h a, v16h b, v8f c) {
  return __builtin_amdgcn_wmma_f32_16x16x32_f16(
      /*neg_a=*/false, a, /*neg_b=*/false, b,
      /*c_mod=*/(short)0, c, /*reuse_a=*/false, /*reuse_b=*/false);
}

// C-layout rows are striped across 16 lanes (N = lane&15) -> reduce within
// each 16-lane half-wave using xor shuffles (masks 1,2,4,8 keep bit4 fixed).
__device__ __forceinline__ float rowmax16(float v) {
  v = fmaxf(v, __shfl_xor(v, 1, 32));
  v = fmaxf(v, __shfl_xor(v, 2, 32));
  v = fmaxf(v, __shfl_xor(v, 4, 32));
  v = fmaxf(v, __shfl_xor(v, 8, 32));
  return v;
}
__device__ __forceinline__ float rowsum16(float v) {
  v += __shfl_xor(v, 1, 32);
  v += __shfl_xor(v, 2, 32);
  v += __shfl_xor(v, 4, 32);
  v += __shfl_xor(v, 8, 32);
  return v;
}

// ---------------------------------------------------------------------------
// TDM: stage a 64(n) x 32(k) f16 tile of BT[N][K] into LDS with row padding
// (pad_interval=3 -> every 16 DWORDs (64B row), pad_amount=3 -> 4 DWORDs),
// giving an 80-byte LDS row stride for bank spreading.  2D descriptor per
// ISA cdna5 ch.8; groups 2/3 (+trailing group) zero: dims 2+ unused.
// This toolchain exposes the 6-arg builtin form.
// ---------------------------------------------------------------------------
__device__ __forceinline__ void tdm_stage_b(const _Float16* gsrc, unsigned lds_off,
                                            int K, int N) {
  unsigned long long ga = (unsigned long long)(uintptr_t)gsrc;
  u32x4 g0;
  g0[0] = 1u;                                        // count=1, user mode
  g0[1] = lds_off;                                   // lds_addr (bytes)
  g0[2] = (unsigned)ga;                              // global_addr[31:0]
  g0[3] = (unsigned)((ga >> 32) & 0x1FFFFFFull) | (2u << 30);  // addr[56:32] | type=2
  i32x8 g1;
  g1[0] = (1 << 16) | (1 << 20) | (3 << 22) | (3 << 25); // data_size=2B, pad_en, interval, amount
  g1[1] = (K & 0xFFFF) << 16;                        // tensor_dim0[15:0]
  g1[2] = ((unsigned)K >> 16) | ((N & 0xFFFF) << 16);// tensor_dim0[31:16] | tensor_dim1[15:0]
  g1[3] = ((unsigned)N >> 16) | (32 << 16);          // tensor_dim1[31:16] | tile_dim0=32
  g1[4] = 64;                                        // tile_dim1=64, tile_dim2=0
  g1[5] = K;                                         // tensor_dim0_stride[31:0]
  g1[6] = 0;                                         // stride hi | dim1_stride lo
  g1[7] = 0;
  i32x4 gz4 = {0, 0, 0, 0};
  i32x8 gz8 = {0, 0, 0, 0, 0, 0, 0, 0};
  __builtin_amdgcn_tensor_load_to_lds(g0, g1, gz4, gz4, gz8, 0);
}

// ---------------------------------------------------------------------------
// f32 -> f16 convert (row-major copy)
// ---------------------------------------------------------------------------
__global__ void convert_f16(const float* __restrict__ in, _Float16* __restrict__ out, int n) {
  int i = blockIdx.x * blockDim.x + threadIdx.x;
  if (i < n) out[i] = (_Float16)in[i];
}

// out[n*K + k] = (f16) in[k*N + n]   (weights stored N-major so B-fragments
// become contiguous K-runs)
__global__ void transpose_f16(const float* __restrict__ in, _Float16* __restrict__ out,
                              int K, int N) {
  int i = blockIdx.x * blockDim.x + threadIdx.x;
  if (i >= K * N) return;
  int n = i / K, k = i % K;
  out[i] = (_Float16)in[(size_t)k * N + n];
}

// ---------------------------------------------------------------------------
// WMMA GEMM with TDM-staged B: C[M,N] f32 = A[M,K] f16 * BT[N,K] f16.
// Block = 8 waves, one 128x64 output tile; all waves share the same 64 N
// columns, so the 64x32 B chunk is staged once per block by the Tensor Data
// Mover (double-buffered, overlapping the DMA of chunk k+1 with WMMAs on k).
// ---------------------------------------------------------------------------
__global__ void __launch_bounds__(256) wmma_gemm_f16(
    const _Float16* __restrict__ A, const _Float16* __restrict__ BT,
    float* __restrict__ C, int M, int N, int K) {
  __shared__ _Float16 bbuf[2][64 * BROW];            // 2 x 5120 B
  const int wave = threadIdx.x >> 5;
  const int lane = threadIdx.x & 31;
  const int l15  = lane & 15;
  const int hib  = lane >> 4;
  const int tiles_m = M >> 4;
  const int gw = blockIdx.x * 8 + wave;
  const int tm = gw % tiles_m;                       // per-wave row tile
  const int tn = gw / tiles_m;                       // block-uniform col tile
  const int m0 = tm << 4, n0 = tn << 6;
  if (n0 >= N) return;                               // block-uniform (never here)

  const unsigned lds0 = (unsigned)(uintptr_t)(&bbuf[0][0]);  // LDS aperture: low 32b = offset
  const unsigned bufbytes = 64 * BROW * 2;
  const int nks = K >> 5;

  if (wave == 0) tdm_stage_b(BT + (size_t)n0 * K, lds0, K, N);

  v8f acc[4] = {};
  const _Float16* arow = A + (size_t)(m0 + l15) * K;

  for (int ks = 0; ks < nks; ++ks) {
    const int k0 = ks << 5;
    if (wave == 0) __builtin_amdgcn_s_wait_tensorcnt(0);
    __syncthreads();                                 // publish buf[ks&1]
    if (wave == 0 && ks + 1 < nks)                   // prefetch next chunk via TDM
      tdm_stage_b(BT + (size_t)n0 * K + (k0 + 32), lds0 + ((ks + 1) & 1) * bufbytes,
                  K, N);

    // A fragment (ISA 16-bit A 16x32 layout): halves 0-7 -> K = k0+8*hib..,
    // halves 8-15 -> K = k0+16+8*hib..
    v16h a;
    {
      v8h lo = *(const v8h*)(arow + k0 + 8 * hib);
      v8h hi = *(const v8h*)(arow + k0 + 16 + 8 * hib);
#pragma unroll
      for (int i = 0; i < 8; ++i) { a[i] = lo[i]; a[8 + i] = hi[i]; }
    }
    const _Float16* bs = &bbuf[ks & 1][0];
#pragma unroll
    for (int j = 0; j < 4; ++j) {
      // B fragment from LDS: lane = N column, halves 0-7 -> K = 8*hib.., halves 8-15 -> K = 16+8*hib..
      const _Float16* brow = bs + (j * 16 + l15) * BROW + 8 * hib;
      v16h b;
      v8h lo = *(const v8h*)(brow);
      v8h hi = *(const v8h*)(brow + 16);
#pragma unroll
      for (int i = 0; i < 8; ++i) { b[i] = lo[i]; b[8 + i] = hi[i]; }
      acc[j] = wmma_f16(a, b, acc[j]);
    }
    __syncthreads();                                 // done reading buf[ks&1]
  }
#pragma unroll
  for (int j = 0; j < 4; ++j)
#pragma unroll
    for (int r = 0; r < 8; ++r)
      C[(size_t)(m0 + r + 8 * hib) * N + (n0 + j * 16 + l15)] = acc[j][r];
}

// ---------------------------------------------------------------------------
// RMSNorm + RoPE + pack into attention-friendly layouts.
// One wave handles one (t, unit) pair; unit 0-15 = q head, 16-17 = k head,
// 18-19 = v head. Each lane owns d = lane + 32*i (i<4); the rotate-half
// partner (d, d+-64) lives in the same lane at element i+-2.
// ---------------------------------------------------------------------------
__global__ void __launch_bounds__(256) norm_rope_pack(
    const float* __restrict__ qkv, const float* __restrict__ cosT,
    const float* __restrict__ sinT, const float* __restrict__ qw,
    const float* __restrict__ kw, _Float16* __restrict__ qh,
    _Float16* __restrict__ kh, _Float16* __restrict__ vT) {
  int wid  = (blockIdx.x * blockDim.x + threadIdx.x) >> 5;
  int lane = threadIdx.x & 31;
  int t = wid / 20, u = wid % 20;
  if (t >= T_DIM) return;

  const float* src;
  if (u < NH)          src = qkv + (size_t)t * NQKV + u * (2 * HD);           // q of head u
  else if (u < NH + 2) src = qkv + (size_t)t * NQKV + QSZ + (u - NH) * HD;    // k
  else                 src = qkv + (size_t)t * NQKV + QSZ + NKV * HD + (u - NH - 2) * HD; // v

  float x[4];
#pragma unroll
  for (int i = 0; i < 4; ++i) x[i] = src[lane + 32 * i];

  if (u < NH + 2) {
    float ss = 0.f;
#pragma unroll
    for (int i = 0; i < 4; ++i) ss += x[i] * x[i];
    for (int m = 1; m < 32; m <<= 1) ss += __shfl_xor(ss, m, 32);
    float r = rsqrtf(ss * (1.0f / HD) + 1e-6f);
    const float* w = (u < NH) ? qw : kw;
    float nx[4];
#pragma unroll
    for (int i = 0; i < 4; ++i) nx[i] = x[i] * r * w[lane + 32 * i];
    float o[4];
#pragma unroll
    for (int i = 0; i < 4; ++i) {
      int d = lane + 32 * i;
      float rot = (i < 2) ? -nx[i + 2] : nx[i - 2];
      o[i] = nx[i] * cosT[(size_t)t * HD + d] + rot * sinT[(size_t)t * HD + d];
    }
    if (u < NH) {
      _Float16* dst = qh + ((size_t)u * T_DIM + t) * HD;
#pragma unroll
      for (int i = 0; i < 4; ++i) dst[lane + 32 * i] = (_Float16)o[i];
    } else {
      _Float16* dst = kh + ((size_t)(u - NH) * T_DIM + t) * HD;
#pragma unroll
      for (int i = 0; i < 4; ++i) dst[lane + 32 * i] = (_Float16)o[i];
    }
  } else {
    int hv = u - NH - 2;   // vT[hkv][d][t]
#pragma unroll
    for (int i = 0; i < 4; ++i) {
      int d = lane + 32 * i;
      vT[((size_t)hv * HD + d) * T_DIM + t] = (_Float16)x[i];
    }
  }
}

// ---------------------------------------------------------------------------
// Flash attention, one wave per (head, 16-query tile). Online softmax over
// 32-key blocks; P converted C-layout -> A-layout via per-wave LDS tile
// (wave-internal cross-lane exchange guarded by s_wait_dscnt).  Sigmoid gate
// fused into the epilogue; emits f16 gctx[T, H*D].
// ---------------------------------------------------------------------------
__global__ void __launch_bounds__(256) flash_attn(
    const _Float16* __restrict__ qh, const _Float16* __restrict__ kh,
    const _Float16* __restrict__ vT, const float* __restrict__ qkv,
    _Float16* __restrict__ gctx) {
  __shared__ _Float16 plds[8 * 16 * 32];     // one 16x32 P tile per wave
  const int wave = threadIdx.x >> 5;
  const int lane = threadIdx.x & 31;
  const int l15  = lane & 15;
  const int hib  = lane >> 4;
  const int gw   = blockIdx.x * 8 + wave;
  const int head = gw >> 8;                  // /(T/16)
  const int qt   = gw & 255;
  const int hkv  = head >> 3;                // H/HKV = 8

  // Q fragments for the 16x128 query tile (4 K-steps of 32)
  v16h qf[4];
  const _Float16* qrow = qh + ((size_t)head * T_DIM + qt * 16 + l15) * HD;
#pragma unroll
  for (int ds = 0; ds < 4; ++ds) {
    v8h lo = *(const v8h*)(qrow + ds * 32 + 16 * hib);
    v8h hi = *(const v8h*)(qrow + ds * 32 + 16 * hib + 8);
#pragma unroll
    for (int i = 0; i < 8; ++i) { qf[ds][i] = lo[i]; qf[ds][8 + i] = hi[i]; }
  }

  v8f o[8] = {};
  float mrow[8], lrow[8];
#pragma unroll
  for (int r = 0; r < 8; ++r) { mrow[r] = -1e30f; lrow[r] = 0.f; }

  const int nkb = (qt + 2) >> 1;             // ceil((qt+1)*16 / 32)
  _Float16* pl = plds + wave * 512;
  const float scale = 0.08838834764831845f;  // 1/sqrt(128)

  for (int kb = 0; kb < nkb; ++kb) {
    const int key0 = kb * 32;
    v8f s0 = {}, s1 = {};
#pragma unroll
    for (int ds = 0; ds < 4; ++ds) {
      const v16h* k0p = (const v16h*)(kh + ((size_t)hkv * T_DIM + key0 + l15) * HD + ds * 32 + 16 * hib);
      const v16h* k1p = (const v16h*)(kh + ((size_t)hkv * T_DIM + key0 + 16 + l15) * HD + ds * 32 + 16 * hib);
      s0 = wmma_f16(qf[ds], *k0p, s0);
      s1 = wmma_f16(qf[ds], *k1p, s1);
    }
#pragma unroll
    for (int r = 0; r < 8; ++r) { s0[r] *= scale; s1[r] *= scale; }

    if (kb == nkb - 1) {                     // causal mask only on the diagonal block
#pragma unroll
      for (int r = 0; r < 8; ++r) {
        int trow = qt * 16 + r + 8 * hib;
        if (key0 + l15 > trow)      s0[r] = -1e30f;
        if (key0 + 16 + l15 > trow) s1[r] = -1e30f;
      }
    }

    float alpha[8];
#pragma unroll
    for (int r = 0; r < 8; ++r) {
      float v  = rowmax16(fmaxf(s0[r], s1[r]));
      float mn = fmaxf(mrow[r], v);
      alpha[r] = __expf(mrow[r] - mn);
      float p0 = __expf(s0[r] - mn);
      float p1 = __expf(s1[r] - mn);
      s0[r] = p0; s1[r] = p1;
      lrow[r] = lrow[r] * alpha[r] + rowsum16(p0 + p1);
      mrow[r] = mn;
    }
#pragma unroll
    for (int dt = 0; dt < 8; ++dt)
#pragma unroll
      for (int r = 0; r < 8; ++r) o[dt][r] *= alpha[r];

    // C-layout P -> LDS [row][klocal]
#pragma unroll
    for (int r = 0; r < 8; ++r) {
      int row = r + 8 * hib;
      pl[row * 32 + l15]      = (_Float16)s0[r];
      pl[row * 32 + 16 + l15] = (_Float16)s1[r];
    }
    asm volatile("s_wait_dscnt 0" ::: "memory");  // wave-internal cross-lane exchange

    v16h pf;                                 // A-layout P fragment
    {
      v8h lo = *(const v8h*)(pl + l15 * 32 + 16 * hib);
      v8h hi = *(const v8h*)(pl + l15 * 32 + 16 * hib + 8);
#pragma unroll
      for (int i = 0; i < 8; ++i) { pf[i] = lo[i]; pf[8 + i] = hi[i]; }
    }
#pragma unroll
    for (int dt = 0; dt < 8; ++dt) {
      const v16h* vp = (const v16h*)(vT + ((size_t)hkv * HD + dt * 16 + l15) * T_DIM + key0 + 16 * hib);
      o[dt] = wmma_f16(pf, *vp, o[dt]);
    }
  }

  // epilogue: 1/l normalization + sigmoid gate, write f16 gated context
#pragma unroll
  for (int r = 0; r < 8; ++r) {
    float inv = 1.0f / lrow[r];
    int t = qt * 16 + r + 8 * hib;
    const float* grow = qkv + (size_t)t * NQKV + head * (2 * HD) + HD;
#pragma unroll
    for (int dt = 0; dt < 8; ++dt) {
      int d = dt * 16 + l15;
      float g   = grow[d];
      float val = o[dt][r] * inv * (1.0f / (1.0f + __expf(-g)));
      gctx[(size_t)t * (NH * HD) + head * HD + d] = (_Float16)val;
    }
  }
}

// ---------------------------------------------------------------------------
extern "C" void kernel_launch(void* const* d_in, const int* in_sizes, int n_in,
                              void* d_out, int out_size, void* d_ws, size_t ws_size,
                              hipStream_t stream) {
  const float* hs   = (const float*)d_in[0];
  const float* cosT = (const float*)d_in[1];
  const float* sinT = (const float*)d_in[2];
  const float* wqkv = (const float*)d_in[3];
  const float* wo   = (const float*)d_in[4];
  const float* qw   = (const float*)d_in[5];
  const float* kw   = (const float*)d_in[6];
  float* out = (float*)d_out;

  char* ws = (char*)d_ws;
  size_t off = 0;
  auto alloc = [&](size_t bytes) -> void* {
    void* p = ws + off;
    off += (bytes + 255) & ~(size_t)255;
    return p;
  };
  _Float16* Xh    = (_Float16*)alloc((size_t)T_DIM * HID_DIM * 2);
  _Float16* WqkvT = (_Float16*)alloc((size_t)NQKV * HID_DIM * 2);
  _Float16* WoT   = (_Float16*)alloc((size_t)HID_DIM * HID_DIM * 2);
  float*    qkvb  = (float*)   alloc((size_t)T_DIM * NQKV * 4);
  _Float16* qh    = (_Float16*)alloc((size_t)NH * T_DIM * HD * 2);
  _Float16* kh    = (_Float16*)alloc((size_t)NKV * T_DIM * HD * 2);
  _Float16* vT    = (_Float16*)alloc((size_t)NKV * HD * T_DIM * 2);
  _Float16* gctx  = (_Float16*)alloc((size_t)T_DIM * NH * HD * 2);

  convert_f16<<<(T_DIM * HID_DIM + 255) / 256, 256, 0, stream>>>(hs, Xh, T_DIM * HID_DIM);
  transpose_f16<<<(HID_DIM * NQKV + 255) / 256, 256, 0, stream>>>(wqkv, WqkvT, HID_DIM, NQKV);
  transpose_f16<<<(HID_DIM * HID_DIM + 255) / 256, 256, 0, stream>>>(wo, WoT, HID_DIM, HID_DIM);

  // qkv = Xh @ Wqkv :  (NQKV/64) * (T/16) wave-tiles, 8 waves/block
  wmma_gemm_f16<<<(NQKV / 64) * (T_DIM / 16) / 8, 256, 0, stream>>>(
      Xh, WqkvT, qkvb, T_DIM, NQKV, HID_DIM);

  norm_rope_pack<<<(T_DIM * 20) / 8, 256, 0, stream>>>(
      qkvb, cosT, sinT, qw, kw, qh, kh, vT);

  flash_attn<<<(NH * (T_DIM / 16)) / 8, 256, 0, stream>>>(qh, kh, vT, qkvb, gctx);

  // out = gctx @ Wo
  wmma_gemm_f16<<<(HID_DIM / 64) * (T_DIM / 16) / 8, 256, 0, stream>>>(
      gctx, WoT, out, T_DIM, HID_DIM, HID_DIM);
}
